// LoraGroupedLinear_23459111371078
// MI455X (gfx1250) — matrix operand order repaired and checked by reading, not compiled, not run on hardware
//
#include <hip/hip_runtime.h>
#include <hip/hip_bf16.h>
#include <stdint.h>

// ---------------------------------------------------------------------------
// LoRA grouped linear for MI455X (gfx1250, wave32, WMMA, async-to-LDS).
// out[T,DOUT] = x @ W_base[e] + SCALE * ((x @ W_a[e]) @ W_b[e])
// fp32 refs; K-reduction on bf16 WMMA pipe with hi/lo split (3 WMMAs per
// tile-pair, f32 accumulate) for fp32-equivalent accuracy.
// ---------------------------------------------------------------------------

#define NUM_EXPERTS 8
#define T_TOK       16384
#define DIM_IN      2048
#define DIM_OUT     2048
#define RANK        64
#define LORA_SCALE  2.0f   // ALPHA / RANK = 128 / 64

typedef __bf16 bf16;
typedef __attribute__((ext_vector_type(4)))  bf16  v4bf;
typedef __attribute__((ext_vector_type(8)))  bf16  v8bf;
typedef __attribute__((ext_vector_type(16))) bf16  v16bf;
typedef __attribute__((ext_vector_type(8)))  float v8f;

// ---- gfx1250 async copy (ASYNCcnt path) ------------------------------------

__device__ __forceinline__ uint32_t lds_addr32(const void* p) {
  // low 32 bits of a generic shared-aperture address == LDS byte offset
  return (uint32_t)(uintptr_t)p;
}

__device__ __forceinline__ void async_copy_b128(void* lds_dst, const void* gsrc) {
  asm volatile("global_load_async_to_lds_b128 %0, %1, off"
               :: "v"(lds_addr32(lds_dst)), "v"(gsrc)
               : "memory");
}

__device__ __forceinline__ void wait_asynccnt0() {
  asm volatile("s_wait_asynccnt 0x0" ::: "memory");
}

// ---- split fp32 -> hi/lo bf16 ----------------------------------------------

__device__ __forceinline__ void split4(float f0, float f1, float f2, float f3,
                                       v4bf& hv, v4bf& lv) {
  const bf16 h0 = (bf16)f0, h1 = (bf16)f1, h2 = (bf16)f2, h3 = (bf16)f3;
  hv.x = h0; hv.y = h1; hv.z = h2; hv.w = h3;
  lv.x = (bf16)(f0 - (float)h0);
  lv.y = (bf16)(f1 - (float)h1);
  lv.z = (bf16)(f2 - (float)h2);
  lv.w = (bf16)(f3 - (float)h3);
}

// ---- WMMA fragments (ISA 7.12.2 lane layouts) ------------------------------

// A fragment 16x32 bf16, LDS row-major ld=32.
__device__ __forceinline__ v16bf frag_a(const bf16* __restrict__ base, int row0) {
  const int lane = threadIdx.x & 31;
  const bf16* p = base + (row0 + (lane & 15)) * 32 + ((lane >> 4) << 3);
  v8bf x0 = *(const v8bf*)(p);       // K 0..7   (or 8..15)
  v8bf x1 = *(const v8bf*)(p + 16);  // K 16..23 (or 24..31)
  return __builtin_shufflevector(x0, x1, 0,1,2,3,4,5,6,7,8,9,10,11,12,13,14,15);
}

// B fragment 32x16 bf16, LDS column-major [n][k] ld=32.
__device__ __forceinline__ v16bf frag_b(const bf16* __restrict__ base, int col0) {
  const int lane = threadIdx.x & 31;
  const bf16* p = base + (col0 + (lane & 15)) * 32 + ((lane >> 4) << 3);
  v8bf x0 = *(const v8bf*)(p);
  v8bf x1 = *(const v8bf*)(p + 16);
  return __builtin_shufflevector(x0, x1, 0,1,2,3,4,5,6,7,8,9,10,11,12,13,14,15);
}

// Split-bf16 MAC: c += a_hi*b_hi + a_lo*b_hi + a_hi*b_lo (f32 accumulate).
__device__ __forceinline__ v8f wmma3(v16bf ah, v16bf al, v16bf bh, v16bf bl, v8f c) {
  c = __builtin_amdgcn_wmma_f32_16x16x32_bf16(false, ah, false, bh, (short)0, c, false, false);
  c = __builtin_amdgcn_wmma_f32_16x16x32_bf16(false, al, false, bh, (short)0, c, false, false);
  c = __builtin_amdgcn_wmma_f32_16x16x32_bf16(false, ah, false, bl, (short)0, c, false, false);
  return c;
}

__device__ __forceinline__ int expert_of(const int* __restrict__ tpe, int row) {
  int acc = 0, e = 0;
#pragma unroll
  for (int i = 0; i < NUM_EXPERTS; ++i) {
    if (row >= acc) e = i;
    acc += tpe[i];
  }
  return e;
}

// ---- register-staged tile loaders / LDS writers ----------------------------

// A tile 128x32 fp32: float4 along K, packed b64 plane stores.
__device__ __forceinline__ void load_a_regs(const float* __restrict__ srcA, int lda,
                                            int m0, int k0, int tid, float4 (&ra)[4]) {
#pragma unroll
  for (int i = 0; i < 4; ++i) {
    const int idx = tid + i * 256;
    const int r = idx >> 3, kq = (idx & 7) << 2;
    ra[i] = *(const float4*)(srcA + (size_t)(m0 + r) * lda + k0 + kq);
  }
}

__device__ __forceinline__ void store_a_lds(bf16* __restrict__ Ah, bf16* __restrict__ Al,
                                            int tid, const float4 (&ra)[4]) {
#pragma unroll
  for (int i = 0; i < 4; ++i) {
    const int idx = tid + i * 256;
    const int r = idx >> 3, kq = (idx & 7) << 2;
    v4bf hv, lv;
    split4(ra[i].x, ra[i].y, ra[i].z, ra[i].w, hv, lv);
    *(v4bf*)(Ah + r * 32 + kq) = hv;
    *(v4bf*)(Al + r * 32 + kq) = lv;
  }
}

// B tile 32xCOLS fp32 row-major: per-thread 4 K-contiguous scalars (coalesced
// across n) -> packed b64 column-major plane stores.
template <int COLS>
__device__ __forceinline__ void load_b_regs(const float* __restrict__ srcB, int ldb,
                                            int n0, int k0, int tid,
                                            float (&rb)[COLS * 8 / 256][4]) {
#pragma unroll
  for (int i = 0; i < COLS * 8 / 256; ++i) {
    const int idx = tid + i * 256;
    const int n = idx & (COLS - 1);
    const int kb = (idx / COLS) << 2;
#pragma unroll
    for (int j = 0; j < 4; ++j)
      rb[i][j] = srcB[(size_t)(k0 + kb + j) * ldb + n0 + n];
  }
}

template <int COLS>
__device__ __forceinline__ void store_b_lds(bf16* __restrict__ Bh, bf16* __restrict__ Bl,
                                            int tid, const float (&rb)[COLS * 8 / 256][4]) {
#pragma unroll
  for (int i = 0; i < COLS * 8 / 256; ++i) {
    const int idx = tid + i * 256;
    const int n = idx & (COLS - 1);
    const int kb = (idx / COLS) << 2;
    v4bf hv, lv;
    split4(rb[i][0], rb[i][1], rb[i][2], rb[i][3], hv, lv);
    *(v4bf*)(Bh + n * 32 + kb) = hv;
    *(v4bf*)(Bl + n * 32 + kb) = lv;
  }
}

// ---------------------------------------------------------------------------
// One-time: transpose+split W_b -> bf16 planes wbT[e][n][k] (hi, lo).
// ---------------------------------------------------------------------------
__global__ __launch_bounds__(256)
void wb_preconvert_kernel(const float* __restrict__ w_b,
                          bf16* __restrict__ wbT_h, bf16* __restrict__ wbT_l) {
  const int idx = blockIdx.x * 256 + threadIdx.x;  // over [e][k][n]
  const int n = idx & (DIM_OUT - 1);
  const int k = (idx >> 11) & (RANK - 1);
  const int e = idx >> 17;
  const float f = w_b[idx];
  const bf16 h = (bf16)f;
  const size_t o = ((size_t)e * DIM_OUT + n) * RANK + k;
  wbT_h[o] = h;
  wbT_l[o] = (bf16)(f - (float)h);
}

// ---------------------------------------------------------------------------
// Kernel 1: mid planes = split_bf16(SCALE * (x @ W_a[e]))   [T, RANK]
// 128x64 tile / block, double-buffered LDS, register pipeline.
// ---------------------------------------------------------------------------
__global__ __launch_bounds__(256)
void lora_mid_kernel(const float* __restrict__ x, const int* __restrict__ tpe,
                     const float* __restrict__ w_a,
                     bf16* __restrict__ mid_h, bf16* __restrict__ mid_l) {
  __shared__ alignas(16) bf16 Ah[2][128 * 32];
  __shared__ alignas(16) bf16 Al[2][128 * 32];
  __shared__ alignas(16) bf16 Bh[2][64 * 32];
  __shared__ alignas(16) bf16 Bl[2][64 * 32];

  const int tid = threadIdx.x;
  const int m0 = blockIdx.x * 128;
  const int e = expert_of(tpe, m0);
  const float* wa = w_a + (size_t)e * DIM_IN * RANK;

  const int warp = tid >> 5;
  const int wm = (warp & 3) * 32;   // 4 M-waves * 32 rows
  const int wn = (warp >> 2) * 32;  // 2 N-waves * 32 cols

  const v8f vzero = {0.f, 0.f, 0.f, 0.f, 0.f, 0.f, 0.f, 0.f};
  v8f acc[2][2];
#pragma unroll
  for (int mi = 0; mi < 2; ++mi)
#pragma unroll
    for (int ni = 0; ni < 2; ++ni) acc[mi][ni] = vzero;

  float4 ra[4];
  float rb[2][4];
  load_a_regs(x, DIM_IN, m0, 0, tid, ra);
  load_b_regs<64>(wa, RANK, 0, 0, tid, rb);

  int buf = 0;
  for (int k0 = 0; k0 < DIM_IN; k0 += 32) {
    store_a_lds(Ah[buf], Al[buf], tid, ra);
    store_b_lds<64>(Bh[buf], Bl[buf], tid, rb);
    if (k0 + 32 < DIM_IN) {
      load_a_regs(x, DIM_IN, m0, k0 + 32, tid, ra);
      load_b_regs<64>(wa, RANK, 0, k0 + 32, tid, rb);
    }
    __syncthreads();
    v16bf bh[2], bl[2];
#pragma unroll
    for (int ni = 0; ni < 2; ++ni) {
      bh[ni] = frag_b(Bh[buf], wn + ni * 16);
      bl[ni] = frag_b(Bl[buf], wn + ni * 16);
    }
#pragma unroll
    for (int mi = 0; mi < 2; ++mi) {
      const v16bf fah = frag_a(Ah[buf], wm + mi * 16);
      const v16bf fal = frag_a(Al[buf], wm + mi * 16);
#pragma unroll
      for (int ni = 0; ni < 2; ++ni)
        acc[mi][ni] = wmma3(fah, fal, bh[ni], bl[ni], acc[mi][ni]);
    }
    buf ^= 1;
  }

  // store scaled mid as bf16 hi/lo planes
  const int lane = tid & 31;
  const int cc = lane & 15, hh = lane >> 4;
#pragma unroll
  for (int mi = 0; mi < 2; ++mi)
#pragma unroll
    for (int ni = 0; ni < 2; ++ni) {
      union { v8f v; float f[8]; } u;
      u.v = acc[mi][ni];
      const int col = wn + ni * 16 + cc;
      const int row0 = m0 + wm + mi * 16 + hh * 8;
#pragma unroll
      for (int v = 0; v < 8; ++v) {
        const float s = LORA_SCALE * u.f[v];
        const bf16 h = (bf16)s;
        const size_t o = (size_t)(row0 + v) * RANK + col;
        mid_h[o] = h;
        mid_l[o] = (bf16)(s - (float)h);
      }
    }
}

// ---------------------------------------------------------------------------
// Kernel 2: out tile 128x128 = x @ W_base[e] (K=2048, split on the fly)
//                            + mid @ W_bT[e] (K=64, async bf16 plane copies)
// ---------------------------------------------------------------------------
__global__ __launch_bounds__(256)
void lora_main_kernel(const float* __restrict__ x, const int* __restrict__ tpe,
                      const float* __restrict__ w_base,
                      const bf16* __restrict__ mid_h, const bf16* __restrict__ mid_l,
                      const bf16* __restrict__ wbT_h, const bf16* __restrict__ wbT_l,
                      float* __restrict__ out) {
  __shared__ alignas(16) bf16 Ah[2][128 * 32];
  __shared__ alignas(16) bf16 Al[2][128 * 32];
  __shared__ alignas(16) bf16 Bh[2][128 * 32];
  __shared__ alignas(16) bf16 Bl[2][128 * 32];

  const int tid = threadIdx.x;
  const int m0 = blockIdx.x * 128;
  const int n0 = blockIdx.y * 128;
  const int e = expert_of(tpe, m0);
  const float* wbase = w_base + (size_t)e * DIM_IN * DIM_OUT;

  const int warp = tid >> 5;
  const int wm = (warp & 3) * 32;   // 4 M-waves * 32 rows
  const int wn = (warp >> 2) * 64;  // 2 N-waves * 64 cols

  const v8f vzero = {0.f, 0.f, 0.f, 0.f, 0.f, 0.f, 0.f, 0.f};
  v8f acc[2][4];
#pragma unroll
  for (int mi = 0; mi < 2; ++mi)
#pragma unroll
    for (int ni = 0; ni < 4; ++ni) acc[mi][ni] = vzero;

  // ---- pass 0: base GEMM, register pipeline + ping-pong LDS ----
  float4 ra[4];
  float rb[4][4];
  load_a_regs(x, DIM_IN, m0, 0, tid, ra);
  load_b_regs<128>(wbase, DIM_OUT, n0, 0, tid, rb);

  int buf = 0;
  for (int k0 = 0; k0 < DIM_IN; k0 += 32) {
    store_a_lds(Ah[buf], Al[buf], tid, ra);
    store_b_lds<128>(Bh[buf], Bl[buf], tid, rb);
    if (k0 + 32 < DIM_IN) {
      load_a_regs(x, DIM_IN, m0, k0 + 32, tid, ra);
      load_b_regs<128>(wbase, DIM_OUT, n0, k0 + 32, tid, rb);
    }
    __syncthreads();
    v16bf bh[4], bl[4];
#pragma unroll
    for (int ni = 0; ni < 4; ++ni) {
      bh[ni] = frag_b(Bh[buf], wn + ni * 16);
      bl[ni] = frag_b(Bl[buf], wn + ni * 16);
    }
#pragma unroll
    for (int mi = 0; mi < 2; ++mi) {
      const v16bf fah = frag_a(Ah[buf], wm + mi * 16);
      const v16bf fal = frag_a(Al[buf], wm + mi * 16);
#pragma unroll
      for (int ni = 0; ni < 4; ++ni)
        acc[mi][ni] = wmma3(fah, fal, bh[ni], bl[ni], acc[mi][ni]);
    }
    buf ^= 1;
  }

  // ---- pass 1: LoRA GEMM, pure bf16 plane copies via async-to-LDS ----
  const bf16* wh = wbT_h + (size_t)e * DIM_OUT * RANK;
  const bf16* wl = wbT_l + (size_t)e * DIM_OUT * RANK;

  for (int k0 = 0; k0 < RANK; k0 += 32) {
    __syncthreads();  // prior compute on this buffer finished everywhere
#pragma unroll
    for (int i = 0; i < 2; ++i) {
      const int idx = tid + i * 256;     // 512 b128 tasks per plane
      const int r = idx >> 2;            // 0..127 (row for A / col for B)
      const int ko = (idx & 3) << 3;     // 0,8,16,24 (bf16 elems)
      async_copy_b128(&Ah[buf][r * 32 + ko], mid_h + (size_t)(m0 + r) * RANK + k0 + ko);
      async_copy_b128(&Al[buf][r * 32 + ko], mid_l + (size_t)(m0 + r) * RANK + k0 + ko);
      async_copy_b128(&Bh[buf][r * 32 + ko], wh + (size_t)(n0 + r) * RANK + k0 + ko);
      async_copy_b128(&Bl[buf][r * 32 + ko], wl + (size_t)(n0 + r) * RANK + k0 + ko);
    }
    wait_asynccnt0();
    __syncthreads();
    v16bf bh[4], bl[4];
#pragma unroll
    for (int ni = 0; ni < 4; ++ni) {
      bh[ni] = frag_b(Bh[buf], wn + ni * 16);
      bl[ni] = frag_b(Bl[buf], wn + ni * 16);
    }
#pragma unroll
    for (int mi = 0; mi < 2; ++mi) {
      const v16bf fah = frag_a(Ah[buf], wm + mi * 16);
      const v16bf fal = frag_a(Al[buf], wm + mi * 16);
#pragma unroll
      for (int ni = 0; ni < 4; ++ni)
        acc[mi][ni] = wmma3(fah, fal, bh[ni], bl[ni], acc[mi][ni]);
    }
    buf ^= 1;
  }

  // ---- store output tile ----
  const int lane = tid & 31;
  const int cc = lane & 15, hh = lane >> 4;
#pragma unroll
  for (int mi = 0; mi < 2; ++mi)
#pragma unroll
    for (int ni = 0; ni < 4; ++ni) {
      union { v8f v; float f[8]; } u;
      u.v = acc[mi][ni];
      const int col = n0 + wn + ni * 16 + cc;
      const int row0 = m0 + wm + mi * 16 + hh * 8;
#pragma unroll
      for (int v = 0; v < 8; ++v)
        out[(size_t)(row0 + v) * DIM_OUT + col] = u.f[v];
    }
}

// ---------------------------------------------------------------------------

extern "C" void kernel_launch(void* const* d_in, const int* in_sizes, int n_in,
                              void* d_out, int out_size, void* d_ws, size_t ws_size,
                              hipStream_t stream) {
  (void)in_sizes; (void)n_in; (void)out_size; (void)ws_size;
  const float* x      = (const float*)d_in[0];
  const int*   tpe    = (const int*)  d_in[1];
  const float* w_base = (const float*)d_in[2];
  const float* w_a    = (const float*)d_in[3];
  const float* w_b    = (const float*)d_in[4];
  float* out = (float*)d_out;

  // workspace layout (bf16 planes): 4 x 2 MB = 8 MB
  bf16* mid_h = (bf16*)d_ws;
  bf16* mid_l = mid_h + (size_t)T_TOK * RANK;
  bf16* wbT_h = mid_l + (size_t)T_TOK * RANK;
  bf16* wbT_l = wbT_h + (size_t)NUM_EXPERTS * DIM_OUT * RANK;

  wb_preconvert_kernel<<<dim3((NUM_EXPERTS * RANK * DIM_OUT) / 256), dim3(256), 0, stream>>>(
      w_b, wbT_h, wbT_l);
  lora_mid_kernel<<<dim3(T_TOK / 128), dim3(256), 0, stream>>>(x, tpe, w_a, mid_h, mid_l);
  lora_main_kernel<<<dim3(T_TOK / 128, DIM_OUT / 128), dim3(256), 0, stream>>>(
      x, tpe, w_base, mid_h, mid_l, wbT_h, wbT_l, out);
}
